// RankOne_MoELinear_38835094290479
// MI455X (gfx1250) — matrix operand order repaired and checked by reading, not compiled, not run on hardware
//
#include <hip/hip_runtime.h>

// ---------------------------------------------------------------------------
// RankOne MoE-Linear for MI455X (gfx1250, wave32, WMMA + async-to-LDS).
//
// out[n,o] = sum_i x[n,i]*w0[o,i] + b0[o] + sum_t dot_t[n]*u[o, idx_t[n]]
// Folded into ONE bf16 GEMM with augmented K = IN + POOLS = 4224:
//   xb[n] = [ bf16(x[n]) | gate vector (dot_t scattered at idx_t) ]
//   wb[o] = [ bf16(w0[o]) | bf16(u[o,:]) ]
// bf16 conversion once -> GEMM working set (~104 MB) is L2-resident (192 MB);
// main loop: GLOBAL_LOAD_ASYNC_TO_LDS_B128 double-buffer feeding
// v_wmma_f32_16x16x32_bf16 at a 1:1 ds_load:wmma ratio.
// ---------------------------------------------------------------------------

#define IN_DIM   4096
#define OUT_DIM  4096
#define POOLS    128
#define KTOT     (IN_DIM + POOLS)   // 4224
#define BK       32                 // == WMMA K for bf16
#define BM       128                // M tile (tokens)
#define BN       256                // N tile (out features)
#define LDS_STRIDE 40               // bf16 elems per LDS row: 80 B = 5*16 B

typedef __bf16 bf16;
typedef __attribute__((ext_vector_type(16))) __bf16 v16bf;
typedef __attribute__((ext_vector_type(4)))  __bf16 v4bf;
typedef __attribute__((ext_vector_type(8)))  float  v8f;
typedef __attribute__((ext_vector_type(4)))  unsigned int u32x4;

union FragU { v16bf v; u32x4 q[2]; };

// ---------------------------------------------------------------------------
// Kernel 1: pack w0 (OUT x IN) and u (OUT x POOLS) into wb (OUT x KTOT) bf16.
// ---------------------------------------------------------------------------
__global__ __launch_bounds__(256) void prep_w_kernel(const float* __restrict__ w0,
                                                     const float* __restrict__ u,
                                                     bf16* __restrict__ wb) {
    const int o = blockIdx.x;
    const float* src = w0 + (size_t)o * IN_DIM;
    bf16* dst = wb + (size_t)o * KTOT;
    for (int i = threadIdx.x * 4; i < IN_DIM; i += 256 * 4) {
        float4 f = *(const float4*)(src + i);
        v4bf b; b.x = (bf16)f.x; b.y = (bf16)f.y; b.z = (bf16)f.z; b.w = (bf16)f.w;
        *(v4bf*)(dst + i) = b;
    }
    const float* su = u + (size_t)o * POOLS;
    for (int p = threadIdx.x * 4; p < POOLS; p += 256 * 4) {
        float4 f = *(const float4*)(su + p);
        v4bf b; b.x = (bf16)f.x; b.y = (bf16)f.y; b.z = (bf16)f.z; b.w = (bf16)f.w;
        *(v4bf*)(dst + IN_DIM + p) = b;
    }
}

// ---------------------------------------------------------------------------
// Kernel 2: pack x (N x IN) into xb (N x KTOT) bf16; zero the gate columns.
// ---------------------------------------------------------------------------
__global__ __launch_bounds__(256) void prep_x_kernel(const float* __restrict__ x,
                                                     bf16* __restrict__ xb) {
    const int n = blockIdx.x;
    const float* src = x + (size_t)n * IN_DIM;
    bf16* dst = xb + (size_t)n * KTOT;
    for (int i = threadIdx.x * 4; i < IN_DIM; i += 256 * 4) {
        float4 f = *(const float4*)(src + i);
        v4bf b; b.x = (bf16)f.x; b.y = (bf16)f.y; b.z = (bf16)f.z; b.w = (bf16)f.w;
        *(v4bf*)(dst + i) = b;
    }
    for (int p = threadIdx.x * 4; p < POOLS; p += 256 * 4) {
        v4bf z; z.x = (bf16)0.f; z.y = (bf16)0.f; z.z = (bf16)0.f; z.w = (bf16)0.f;
        *(v4bf*)(dst + IN_DIM + p) = z;
    }
}

// ---------------------------------------------------------------------------
// Kernel 3: router. 16 tokens per 256-thread block.
//  - logits[n,p] = x[n] . wr[p]   (wr = 2 MB, L2-resident)
//  - top-2 per token, dot_t = x[n] . svh[idx_t], scatter bf16(dot) into xb.
// ---------------------------------------------------------------------------
__global__ __launch_bounds__(256) void router_kernel(const float* __restrict__ x,
                                                     const float* __restrict__ wr,
                                                     const float* __restrict__ svh,
                                                     bf16* __restrict__ xb) {
    const int tok0 = blockIdx.x * 16;
    const int tid  = threadIdx.x;

    __shared__ float xs[16][512];   // staged x chunk (32 KB)
    __shared__ float lg[16][128];   // logits (8 KB)
    __shared__ int   sidx[16][2];
    __shared__ float sred[2][16][16];

    const int p    = tid & 127;     // pool id
    const int half = tid >> 7;      // 0: tokens 0..7, 1: tokens 8..15
    float acc[8];
#pragma unroll
    for (int m = 0; m < 8; ++m) acc[m] = 0.f;

    for (int kc = 0; kc < IN_DIM; kc += 512) {
        for (int j = tid * 4; j < 16 * 512; j += 256 * 4) {
            int r = j >> 9, c = j & 511;
            *(float4*)&xs[r][c] = *(const float4*)&x[(size_t)(tok0 + r) * IN_DIM + kc + c];
        }
        __syncthreads();
        const float* wrow = wr + (size_t)p * IN_DIM + kc;
        for (int i = 0; i < 512; i += 4) {
            float4 w = *(const float4*)&wrow[i];
#pragma unroll
            for (int m = 0; m < 8; ++m) {
                const float* xr = &xs[half * 8 + m][i];
                acc[m] += w.x * xr[0] + w.y * xr[1] + w.z * xr[2] + w.w * xr[3];
            }
        }
        __syncthreads();
    }
#pragma unroll
    for (int m = 0; m < 8; ++m) lg[half * 8 + m][p] = acc[m];
    __syncthreads();

    // top-2 scan, one thread per token
    if (tid < 16) {
        float v0 = -3.4e38f, v1 = -3.4e38f;
        int b0i = 0, b1i = 0;
        for (int q = 0; q < 128; ++q) {
            float v = lg[tid][q];
            if (v > v0)      { v1 = v0; b1i = b0i; v0 = v; b0i = q; }
            else if (v > v1) { v1 = v;  b1i = q; }
        }
        sidx[tid][0] = b0i;
        sidx[tid][1] = b1i;
    }
    __syncthreads();

    // cooperative dots: 16 threads per token
    {
        const int m = tid >> 4;
        const int l = tid & 15;
        const int i0 = sidx[m][0], i1 = sidx[m][1];
        const float* xrow = x   + (size_t)(tok0 + m) * IN_DIM;
        const float* s0   = svh + (size_t)i0 * IN_DIM;
        const float* s1   = svh + (size_t)i1 * IN_DIM;
        float d0 = 0.f, d1 = 0.f;
        for (int i = l * 4; i < IN_DIM; i += 64) {
            float4 xv = *(const float4*)&xrow[i];
            float4 a  = *(const float4*)&s0[i];
            float4 b  = *(const float4*)&s1[i];
            d0 += xv.x * a.x + xv.y * a.y + xv.z * a.z + xv.w * a.w;
            d1 += xv.x * b.x + xv.y * b.y + xv.z * b.z + xv.w * b.w;
        }
        sred[0][m][l] = d0;
        sred[1][m][l] = d1;
    }
    __syncthreads();

    if (tid < 16) {
        float d0 = 0.f, d1 = 0.f;
#pragma unroll
        for (int l = 0; l < 16; ++l) { d0 += sred[0][tid][l]; d1 += sred[1][tid][l]; }
        bf16* gate = xb + (size_t)(tok0 + tid) * KTOT + IN_DIM;
        gate[sidx[tid][0]] = (bf16)d0;
        gate[sidx[tid][1]] = (bf16)d1;
    }
}

// ---------------------------------------------------------------------------
// Kernel 4: main bf16 WMMA GEMM with GLOBAL_LOAD_ASYNC_TO_LDS double-buffering.
// out[M=8192, N=4096] = xb[M,KTOT] * wb[N,KTOT]^T + b0
// Block tile 128x256, 8 waves (2 M x 4 N), wave tile 64x64 = 4x4 WMMA accs.
// LDS rows padded to 40 bf16 (80 B = 5 * 16 B): 16B-aligned, bank-conflict-free.
// NOTE: async-to-LDS offset:N is added to BOTH the LDS and global address
// (ISA 08_async_tensor §4.4), so one address pair covers a contiguous row.
// ---------------------------------------------------------------------------
__global__ __launch_bounds__(256) void gemm_wmma_kernel(const bf16* __restrict__ xb,
                                                        const bf16* __restrict__ wb,
                                                        const float* __restrict__ b0,
                                                        float* __restrict__ out) {
    const int tid  = threadIdx.x;
    const int lane = tid & 31;
    const int wave = tid >> 5;
    const int wm   = (wave & 1) * 64;   // wave M offset in tile
    const int wn   = (wave >> 1) * 64;  // wave N offset in tile

    const int tileN = blockIdx.x * BN;
    const int tileM = blockIdx.y * BM;

    __shared__ bf16 As[2][BM][LDS_STRIDE];   // 2 x 10 KB
    __shared__ bf16 Bs[2][BN][LDS_STRIDE];   // 2 x 20 KB

    const unsigned asBase = (unsigned)(size_t)&As[0][0][0];
    const unsigned bsBase = (unsigned)(size_t)&Bs[0][0][0];
    const unsigned aBufB  = (unsigned)(BM * LDS_STRIDE * sizeof(bf16));
    const unsigned bBufB  = (unsigned)(BN * LDS_STRIDE * sizeof(bf16));

    // A staging: thread -> (row tid>>1, 32-byte half-row); 128 rows x 64 B.
    const int aRow = tid >> 1;
    const int aK   = (tid & 1) * 16;          // bf16 index 0 or 16
    const bf16* gA = xb + (size_t)(tileM + aRow) * KTOT + aK;
    // B staging: thread -> full 64-byte row tid; 256 rows x 64 B.
    const int bRow = tid;
    const bf16* gB = wb + (size_t)(tileN + bRow) * KTOT;

    v8f acc[4][4] = {};

    // ---- async issue of one K-tile into LDS buffer `buf` -------------------
    auto issue_tile = [&](int kt, int buf) {
        const unsigned la = asBase + buf * aBufB
                          + (unsigned)(aRow * LDS_STRIDE + aK) * sizeof(bf16);
        const bf16* ga = gA + (size_t)kt * BK;
        asm volatile(
            "global_load_async_to_lds_b128 %0, %1, off\n\t"
            "global_load_async_to_lds_b128 %0, %1, off offset:16"
            :: "v"(la), "v"(ga) : "memory");
        const unsigned lb = bsBase + buf * bBufB
                          + (unsigned)(bRow * LDS_STRIDE) * sizeof(bf16);
        const bf16* gb = gB + (size_t)kt * BK;
        asm volatile(
            "global_load_async_to_lds_b128 %0, %1, off\n\t"
            "global_load_async_to_lds_b128 %0, %1, off offset:16\n\t"
            "global_load_async_to_lds_b128 %0, %1, off offset:32\n\t"
            "global_load_async_to_lds_b128 %0, %1, off offset:48"
            :: "v"(lb), "v"(gb) : "memory");
    };

    issue_tile(0, 0);

    const int NT = KTOT / BK;   // 132
    for (int kt = 0; kt < NT; ++kt) {
        const int buf = kt & 1;

        // tile kt complete for this wave, then for the whole workgroup
        asm volatile("s_wait_asynccnt 0x0" ::: "memory");
        __syncthreads();

        if (kt + 1 < NT) {
            issue_tile(kt + 1, buf ^ 1);
            if (kt + 2 < NT) {
                __builtin_prefetch((const void*)(gA + (size_t)(kt + 2) * BK), 0, 1);
                __builtin_prefetch((const void*)(gB + (size_t)(kt + 2) * BK), 0, 1);
            }
        }

        // A fragments: 16x32 bf16. lanes 0-15: rows, K 0..7 & 16..23;
        // lanes 16-31: same rows, K 8..15 & 24..31 (ISA 16-bit A layout).
        FragU afrag[4];
#pragma unroll
        for (int f = 0; f < 4; ++f) {
            const bf16* pa = &As[buf][wm + f * 16 + (lane & 15)][(lane >> 4) * 8];
            afrag[f].q[0] = *(const u32x4*)(pa);
            afrag[f].q[1] = *(const u32x4*)(pa + 16);
        }
        // B fragments: 32x16 bf16. lane holds column N=lane%16;
        // lanes 0-15 K 0..7 & 16..23, lanes 16-31 K 8..15 & 24..31 (matches A).
        FragU bfrag[4];
#pragma unroll
        for (int f = 0; f < 4; ++f) {
            const bf16* pb = &Bs[buf][wn + f * 16 + (lane & 15)][(lane >> 4) * 8];
            bfrag[f].q[0] = *(const u32x4*)(pb);
            bfrag[f].q[1] = *(const u32x4*)(pb + 16);
        }

#pragma unroll
        for (int fm = 0; fm < 4; ++fm)
#pragma unroll
            for (int fn = 0; fn < 4; ++fn)
                acc[fm][fn] = __builtin_amdgcn_wmma_f32_16x16x32_bf16(
                    /*neg_a=*/false, afrag[fm].v,
                    /*neg_b=*/false, bfrag[fn].v,
                    /*c_mod=*/(short)0, acc[fm][fn],
                    /*reuse_a=*/false, /*reuse_b=*/false);
        // WAR on buf is safe: every frag ds_load above is consumed (dscnt wait)
        // by a WMMA before this wave reaches the next barrier, and writes to
        // this buffer are only issued after that barrier.
    }

    // Epilogue: D layout — VGPR r, lanes 0-15: M=r, N=lane; lanes 16-31: M=r+8.
    const int cl    = lane & 15;
    const int rbase = tileM + wm + ((lane >> 4) << 3);
#pragma unroll
    for (int fn = 0; fn < 4; ++fn) {
        const int col = tileN + wn + fn * 16 + cl;
        const float bias = b0[col];
#pragma unroll
        for (int fm = 0; fm < 4; ++fm) {
#pragma unroll
            for (int r = 0; r < 8; ++r) {
                const int row = rbase + fm * 16 + r;
                out[(size_t)row * OUT_DIM + col] = acc[fm][fn][r] + bias;
            }
        }
    }
}

// ---------------------------------------------------------------------------
// Launcher. Inputs: hidden_states, w0, b0, wr, u, svh, top_k.
// Workspace layout: xb (ntok*KTOT bf16) | wb (OUT*KTOT bf16)  (~104 MB).
// ---------------------------------------------------------------------------
extern "C" void kernel_launch(void* const* d_in, const int* in_sizes, int n_in,
                              void* d_out, int out_size, void* d_ws, size_t ws_size,
                              hipStream_t stream) {
    const float* x   = (const float*)d_in[0];
    const float* w0  = (const float*)d_in[1];
    const float* b0  = (const float*)d_in[2];
    const float* wr  = (const float*)d_in[3];
    const float* u   = (const float*)d_in[4];
    const float* svh = (const float*)d_in[5];
    // d_in[6] = top_k (hardcoded 2 in router)

    const int ntok = in_sizes[0] / IN_DIM;   // 8192

    size_t xb_bytes = (size_t)ntok * KTOT * sizeof(bf16);
    xb_bytes = (xb_bytes + 255) & ~(size_t)255;
    bf16* xb = (bf16*)d_ws;
    bf16* wb = (bf16*)((char*)d_ws + xb_bytes);

    float* out = (float*)d_out;

    prep_x_kernel<<<ntok, 256, 0, stream>>>(x, xb);
    prep_w_kernel<<<OUT_DIM, 256, 0, stream>>>(w0, u, wb);
    router_kernel<<<ntok / 16, 256, 0, stream>>>(x, wr, svh, xb);

    dim3 grid(OUT_DIM / BN, ntok / BM);
    gemm_wmma_kernel<<<grid, 256, 0, stream>>>(xb, wb, b0, out);
}
